// GPT2_68186900791678
// MI455X (gfx1250) — compile-verified, matrix operand order checked by reading, not run on hardware
//
#include <hip/hip_runtime.h>

#define L_  12
#define B_  4
#define S_  1024
#define V_  50257
#define H_  768
#define NH_ 12
#define HD_ 64
#define H3_ (3 * H_)
#define H4_ (4 * H_)
#define BS_ (B_ * S_)

typedef unsigned int u32;
typedef __bf16 bf16x2 __attribute__((ext_vector_type(2)));
typedef __bf16 bf16x8 __attribute__((ext_vector_type(8)));
typedef __bf16 v16bf  __attribute__((ext_vector_type(16)));
typedef float  v8f    __attribute__((ext_vector_type(8)));
typedef float  f32x4  __attribute__((ext_vector_type(4)));
typedef u32    u32x4  __attribute__((ext_vector_type(4)));
typedef u32    u32x8  __attribute__((ext_vector_type(8)));

// ---- WMMA helpers (CDNA5 bf16 16x16x32, f32 accum) -------------------------

__device__ __forceinline__ v16bf mk_frag(bf16x8 lo, bf16x8 hi) {
  v16bf r;
#pragma unroll
  for (int j = 0; j < 8; ++j) { r[j] = lo[j]; r[8 + j] = hi[j]; }
  return r;
}

// A/B 16-bit fragment: per documented layout, each lane holds two contiguous
// 8-element runs:  run0 @ (hi?8:0), run1 @ 16+(hi?8:0)  (relative to k-base).
__device__ __forceinline__ v16bf frag_from(const __bf16* row, int hiOff) {
  bf16x8 r0 = *(const bf16x8*)(row + hiOff);
  bf16x8 r1 = *(const bf16x8*)(row + 16 + hiOff);
  return mk_frag(r0, r1);
}

// Same fragment, but source row is f32 in LDS (TDM-loaded); convert in regs.
__device__ __forceinline__ v16bf frag_from_f32(const float* row, int hiOff) {
  const f32x4* p0 = (const f32x4*)(row + hiOff);
  const f32x4* p1 = (const f32x4*)(row + 16 + hiOff);
  f32x4 a0 = p0[0], a1 = p0[1], b0 = p1[0], b1 = p1[1];
  v16bf r;
#pragma unroll
  for (int j = 0; j < 4; ++j) {
    r[j]      = (__bf16)a0[j];
    r[4 + j]  = (__bf16)a1[j];
    r[8 + j]  = (__bf16)b0[j];
    r[12 + j] = (__bf16)b1[j];
  }
  return r;
}

__device__ __forceinline__ v8f wmma_bf16(v16bf a, v16bf b, v8f c) {
  return __builtin_amdgcn_wmma_f32_16x16x32_bf16(false, a, false, b,
                                                 (short)0, c, false, false);
}

__device__ __forceinline__ bf16x2 pk_bf2(float a, float b) {
  bf16x2 r; r[0] = (__bf16)a; r[1] = (__bf16)b; return r;
}

// ---- Tensor Data Mover: 2D f32 tile (tile_w x tile_h) global -> LDS --------
// D# per CDNA5 ISA. LDS rows get +8 DWORD padding every
// 32 DWORDs (pad_interval code 4, pad_amount code 7) => stride-40-f32 rows.
// Must be called wave-uniformly; tracked with TENSORcnt.

__device__ __forceinline__ void tdm_load_2d_f32(u32 lds_off, const float* gptr,
                                                u32 tile_w, u32 tile_h,
                                                u32 tensor_w, u32 tensor_h,
                                                u32 row_stride_elems) {
  unsigned long long ga = (unsigned long long)(uintptr_t)gptr;
  u32x4 g0;
  g0[0] = 1u;                                   // count=1 (valid user D#)
  g0[1] = lds_off;                              // lds_addr (bytes)
  g0[2] = (u32)ga;                              // global_addr[31:0]
  g0[3] = (u32)((ga >> 32) & 0x1FFFFFFu)        // global_addr[56:32]
        | (2u << 30);                           // type=2 ("image")
  u32x8 g1;
  g1[0] = (2u << 16)                            // data_size=2 (4 bytes)
        | (1u << 20)                            // pad_enable
        | (4u << 22)                            // pad_interval: 32 DWORDs
        | (7u << 25);                           // pad_amount:   8 DWORDs
  g1[1] = (tensor_w & 0xFFFFu) << 16;                        // dim0[15:0]
  g1[2] = (tensor_w >> 16) | ((tensor_h & 0xFFFFu) << 16);   // dim0[31:16],dim1[15:0]
  g1[3] = (tensor_h >> 16) | (tile_w << 16);                 // dim1[31:16],tile_dim0
  g1[4] = tile_h & 0xFFFFu;                                  // tile_dim1 (tile_dim2=0)
  g1[5] = row_stride_elems;                                  // dim0_stride[31:0]
  g1[6] = 0u;                                                // dim0_stride[47:32]
  g1[7] = 0u;
  asm volatile("tensor_load_to_lds %0, %1" :: "s"(g0), "s"(g1) : "memory");
}

// ---- Embedding: x = tok_emb[ids] + pos_emb ---------------------------------

__global__ void k_embed(const int* __restrict__ ids,
                        const float* __restrict__ tok,
                        const float* __restrict__ pos,
                        float* __restrict__ x) {
  int row = blockIdx.x;            // b*S + s
  int s = row & (S_ - 1);
  int id = ids[row];
  for (int h = threadIdx.x; h < H_; h += blockDim.x)
    x[(size_t)row * H_ + h] = tok[(size_t)id * H_ + h] + pos[(size_t)s * H_ + h];
}

// ---- LayerNorm (one 256-thread block per row of H=768) ---------------------

__global__ __launch_bounds__(256) void k_ln(const float* __restrict__ in,
                                            const float* __restrict__ w,
                                            const float* __restrict__ b,
                                            float* __restrict__ out) {
  __shared__ float red[256];
  int row = blockIdx.x;
  int tid = threadIdx.x;
  const float* p = in + (size_t)row * H_;

  float s = 0.f;
  for (int h = tid; h < H_; h += 256) s += p[h];
  red[tid] = s; __syncthreads();
  for (int off = 128; off > 0; off >>= 1) {
    if (tid < off) red[tid] += red[tid + off];
    __syncthreads();
  }
  float mean = red[0] * (1.0f / H_);
  __syncthreads();

  float v = 0.f;
  for (int h = tid; h < H_; h += 256) { float d = p[h] - mean; v += d * d; }
  red[tid] = v; __syncthreads();
  for (int off = 128; off > 0; off >>= 1) {
    if (tid < off) red[tid] += red[tid + off];
    __syncthreads();
  }
  float inv = rsqrtf(red[0] * (1.0f / H_) + 1e-5f);

  for (int h = tid; h < H_; h += 256)
    out[(size_t)row * H_ + h] = (p[h] - mean) * inv * w[h] + b[h];
}

// ---- Tiled WMMA GEMM: C[M,N] = act(A[M,K] @ W[K,N] + bias) (+ resid) -------
// Block: 256 thr = 8 waves; block tile 128(M) x 64(N); K-step 32.
// A tile: raw f32 via Tensor Data Mover (pad-on-store => stride 40).
// W tile: staged transposed (N x K) as bf16 with packed cvt stores.

__global__ __launch_bounds__(256) void k_gemm(const float* __restrict__ A,
                                              const float* __restrict__ W,
                                              const float* __restrict__ bias,
                                              const float* __restrict__ resid,
                                              float* __restrict__ C,
                                              int M, int N, int K, int act) {
  __shared__ __align__(16) float  Asf[128 * 40];  // f32, TDM-padded rows
  __shared__ __align__(16) __bf16 Bs[64 * 40];    // bf16, transposed (N x K)

  int t = threadIdx.x;
  int n0 = blockIdx.x * 64;
  int m0 = blockIdx.y * 128;
  int lane = t & 31, wid = t >> 5;
  int hiOff = (lane & 16) ? 8 : 0;
  int ln = lane & 15;
  int wm = (wid & 3) * 32;   // 4 waves along M
  int wn = (wid >> 2) * 32;  // 2 waves along N

  u32 ldsA = (u32)(uintptr_t)&Asf[0];
  v8f acc[2][2] = {};

  for (int kk = 0; kk < K; kk += 32) {
    // Async DMA: A tile 128 x 32 f32 starting at A[m0][kk]
    if (wid == 0)
      tdm_load_2d_f32(ldsA, A + (size_t)m0 * K + kk,
                      /*tile_w=*/32, /*tile_h=*/128,
                      /*tensor_w=*/(u32)(K - kk), /*tensor_h=*/(u32)(M - m0),
                      /*stride=*/(u32)K);

    // Meanwhile: stage W tile 32x64 -> Bs (transposed), packed bf16 pairs
#pragma unroll
    for (int i = 0; i < 4; ++i) {
      int idx = i * 256 + t;          // 1024 k-pairs
      int n = idx & 63, rp = idx >> 6;
      float w0 = W[(size_t)(kk + 2 * rp) * N + n0 + n];
      float w1 = W[(size_t)(kk + 2 * rp + 1) * N + n0 + n];
      *(bf16x2*)&Bs[n * 40 + 2 * rp] = pk_bf2(w0, w1);
    }

    if (wid == 0) __builtin_amdgcn_s_wait_tensorcnt(0);
    __syncthreads();

    v16bf a0 = frag_from_f32(&Asf[(wm + ln) * 40], hiOff);
    v16bf a1 = frag_from_f32(&Asf[(wm + 16 + ln) * 40], hiOff);
    v16bf b0 = frag_from(&Bs[(wn + ln) * 40], hiOff);
    v16bf b1 = frag_from(&Bs[(wn + 16 + ln) * 40], hiOff);

    acc[0][0] = wmma_bf16(a0, b0, acc[0][0]);
    acc[0][1] = wmma_bf16(a0, b1, acc[0][1]);
    acc[1][0] = wmma_bf16(a1, b0, acc[1][0]);
    acc[1][1] = wmma_bf16(a1, b1, acc[1][1]);
    __syncthreads();
  }

  int hi8 = (lane & 16) ? 8 : 0;
#pragma unroll
  for (int m2 = 0; m2 < 2; ++m2)
#pragma unroll
    for (int n2 = 0; n2 < 2; ++n2)
#pragma unroll
      for (int r = 0; r < 8; ++r) {
        int row = m0 + wm + m2 * 16 + r + hi8;
        int col = n0 + wn + n2 * 16 + ln;
        float v = acc[m2][n2][r] + bias[col];
        if (act == 1)  // exact GELU
          v = 0.5f * v * (1.0f + erff(v * 0.70710678118654752f));
        if (resid) v += resid[(size_t)row * N + col];
        C[(size_t)row * N + col] = v;
      }
}

// ---- Flash attention: one wave32 per (b, head, 16-row q tile) --------------
// qkv is (B,S,3H) f32; o is (B,S,H) f32. Causal, scale folded into Q.

__global__ __launch_bounds__(32) void k_attn(const float* __restrict__ qkv,
                                             float* __restrict__ o) {
  __shared__ __align__(16) __bf16 Ks[32 * 64];  // k rows x d      (row-major)
  __shared__ __align__(16) __bf16 Vs[64 * 32];  // d x k rows      (transposed)
  __shared__ __align__(16) __bf16 Ps[16 * 32];  // probs, q x k

  int qt = blockIdx.x, head = blockIdx.y, b = blockIdx.z;
  int lane = threadIdx.x;
  int hi = lane >> 4;
  int hiOff = hi * 8;
  int ln = lane & 15;
  int qrow0 = qt * 16;

  // Q fragments (scale 1/sqrt(64) folded in)
  const float* qp = qkv + (size_t)(b * S_ + qrow0 + ln) * H3_ + head * HD_;
  v16bf aq[2];
#pragma unroll
  for (int dd = 0; dd < 2; ++dd) {
    v16bf a;
#pragma unroll
    for (int j = 0; j < 8; ++j) {
      a[j]     = (__bf16)(qp[dd * 32 + hiOff + j] * 0.125f);
      a[8 + j] = (__bf16)(qp[dd * 32 + 16 + hiOff + j] * 0.125f);
    }
    aq[dd] = a;
  }

  float m[8], l[8];
  v8f Ot[4] = {};
#pragma unroll
  for (int r = 0; r < 8; ++r) { m[r] = -3.0e38f; l[r] = 0.f; }

  int kend = qrow0 + 16;
  for (int kc = 0; kc < kend; kc += 32) {
    // stage K (row-major, packed d-pairs) and V (transposed, packed k-pairs)
#pragma unroll 4
    for (int i = 0; i < 32; ++i) {
      int idx = i * 32 + lane;        // 1024 d-pairs of K
      int r = idx >> 5, c2 = idx & 31;
      size_t base = (size_t)(b * S_ + kc + r) * H3_ + head * HD_ + 2 * c2 + H_;
      *(bf16x2*)&Ks[r * 64 + 2 * c2] = pk_bf2(qkv[base], qkv[base + 1]);
    }
#pragma unroll 4
    for (int i = 0; i < 32; ++i) {
      int idx = i * 32 + lane;        // 1024 k-pairs of V
      int c = idx & 63, rp = idx >> 6;
      size_t base = (size_t)(b * S_ + kc + 2 * rp) * H3_ + head * HD_ + 2 * H_ + c;
      *(bf16x2*)&Vs[c * 32 + 2 * rp] = pk_bf2(qkv[base], qkv[base + H3_]);
    }

    // scores: Q(16x64) @ K^T -> two 16x16 tiles
    v8f c0 = {}, c1 = {};
#pragma unroll
    for (int dd = 0; dd < 2; ++dd) {
      c0 = wmma_bf16(aq[dd], frag_from(&Ks[ln * 64 + dd * 32], hiOff), c0);
      c1 = wmma_bf16(aq[dd], frag_from(&Ks[(16 + ln) * 64 + dd * 32], hiOff), c1);
    }

    // online softmax (stats per row: VGPR r -> row qrow0 + hi*8 + r)
#pragma unroll
    for (int r = 0; r < 8; ++r) {
      int qr = qrow0 + hi * 8 + r;
      float s0 = (qr < kc + ln)      ? -10000.f : c0[r];
      float s1 = (qr < kc + 16 + ln) ? -10000.f : c1[r];
      float tm = fmaxf(s0, s1);
      tm = fmaxf(tm, __shfl_xor(tm, 1, 16));
      tm = fmaxf(tm, __shfl_xor(tm, 2, 16));
      tm = fmaxf(tm, __shfl_xor(tm, 4, 16));
      tm = fmaxf(tm, __shfl_xor(tm, 8, 16));
      float mn = fmaxf(m[r], tm);
      float alpha = expf(m[r] - mn);
      float p0 = expf(s0 - mn), p1 = expf(s1 - mn);
      float rs = p0 + p1;
      rs += __shfl_xor(rs, 1, 16);
      rs += __shfl_xor(rs, 2, 16);
      rs += __shfl_xor(rs, 4, 16);
      rs += __shfl_xor(rs, 8, 16);
      l[r] = l[r] * alpha + rs;
      m[r] = mn;
#pragma unroll
      for (int n = 0; n < 4; ++n) Ot[n][r] *= alpha;
      Ps[(hi * 8 + r) * 32 + ln]      = (__bf16)p0;
      Ps[(hi * 8 + r) * 32 + 16 + ln] = (__bf16)p1;
    }

    // O += P(16x32) @ V(32x64)
    v16bf ap = frag_from(&Ps[ln * 32], hiOff);
#pragma unroll
    for (int n = 0; n < 4; ++n)
      Ot[n] = wmma_bf16(ap, frag_from(&Vs[(n * 16 + ln) * 32], hiOff), Ot[n]);
  }

#pragma unroll
  for (int r = 0; r < 8; ++r) {
    float inv = 1.0f / l[r];
    size_t row = (size_t)(b * S_ + qrow0 + hi * 8 + r);
#pragma unroll
    for (int n = 0; n < 4; ++n)
      o[row * H_ + head * HD_ + n * 16 + ln] = Ot[n][r] * inv;
  }
}

// ---- Last-token logits + enc_last ------------------------------------------

__global__ void k_logits(const float* __restrict__ enc,
                         const float* __restrict__ tok,
                         float* __restrict__ out) {
  int g = blockIdx.x * blockDim.x + threadIdx.x;
  if (g >= B_ * V_) return;
  int b = g / V_, v = g % V_;
  const float* e = enc + (size_t)(b * S_ + S_ - 1) * H_;
  const float* w = tok + (size_t)v * H_;
  float s = 0.f;
  for (int h = 0; h < H_; ++h) s += e[h] * w[h];
  out[g] = s;
}

__global__ void k_enclast(const float* __restrict__ enc,
                          float* __restrict__ out) {
  int g = blockIdx.x * blockDim.x + threadIdx.x;
  if (g >= B_ * H_) return;
  int b = g / H_, h = g % H_;
  out[g] = enc[(size_t)(b * S_ + S_ - 1) * H_ + h];
}

// ---- Host orchestration -----------------------------------------------------

extern "C" void kernel_launch(void* const* d_in, const int* in_sizes, int n_in,
                              void* d_out, int out_size, void* d_ws, size_t ws_size,
                              hipStream_t stream) {
  (void)in_sizes; (void)n_in; (void)out_size; (void)ws_size;

  const int*   ids  = (const int*)  d_in[0];
  const float* tok  = (const float*)d_in[1];
  const float* pos  = (const float*)d_in[2];
  const float* ln1w = (const float*)d_in[3];
  const float* ln1b = (const float*)d_in[4];
  const float* qkvw = (const float*)d_in[5];
  const float* qkvb = (const float*)d_in[6];
  const float* outw = (const float*)d_in[7];
  const float* outb = (const float*)d_in[8];
  const float* ln2w = (const float*)d_in[9];
  const float* ln2b = (const float*)d_in[10];
  const float* fc1w = (const float*)d_in[11];
  const float* fc1b = (const float*)d_in[12];
  const float* fc2w = (const float*)d_in[13];
  const float* fc2b = (const float*)d_in[14];
  const float* lnfw = (const float*)d_in[15];
  const float* lnfb = (const float*)d_in[16];

  const size_t XN = (size_t)BS_ * H_;      // 3,145,728 floats
  float* ws  = (float*)d_ws;
  float* x   = ws;                          // residual stream
  float* h   = ws + XN;                     // LN output / enc
  float* o   = ws + 2 * XN;                 // attention output
  float* big = ws + 3 * XN;                 // qkv (BS x 3H) aliased w/ fc1 (BS x 4H)
  float* qkv = big;
  float* f1  = big;

  dim3 blk256(256), blk32(32);

  k_embed<<<BS_, blk256, 0, stream>>>(ids, tok, pos, x);

  for (int i = 0; i < L_; ++i) {
    k_ln<<<BS_, blk256, 0, stream>>>(x, ln1w + (size_t)i * H_, ln1b + (size_t)i * H_, h);

    k_gemm<<<dim3(H3_ / 64, BS_ / 128), blk256, 0, stream>>>(
        h, qkvw + (size_t)i * H_ * H3_, qkvb + (size_t)i * H3_, nullptr, qkv,
        BS_, H3_, H_, 0);

    k_attn<<<dim3(S_ / 16, NH_, B_), blk32, 0, stream>>>(qkv, o);

    k_gemm<<<dim3(H_ / 64, BS_ / 128), blk256, 0, stream>>>(
        o, outw + (size_t)i * H_ * H_, outb + (size_t)i * H_, x, x,
        BS_, H_, H_, 0);

    k_ln<<<BS_, blk256, 0, stream>>>(x, ln2w + (size_t)i * H_, ln2b + (size_t)i * H_, h);

    k_gemm<<<dim3(H4_ / 64, BS_ / 128), blk256, 0, stream>>>(
        h, fc1w + (size_t)i * H_ * H4_, fc1b + (size_t)i * H4_, nullptr, f1,
        BS_, H4_, H_, 1);

    k_gemm<<<dim3(H_ / 64, BS_ / 128), blk256, 0, stream>>>(
        f1, fc2w + (size_t)i * H4_ * H_, fc2b + (size_t)i * H_, x, x,
        BS_, H_, H4_, 0);
  }

  k_ln<<<BS_, blk256, 0, stream>>>(x, lnfw, lnfb, h);

  float* out = (float*)d_out;
  k_logits <<<(B_ * V_ + 255) / 256, blk256, 0, stream>>>(h, tok, out);
  k_enclast<<<(B_ * H_ + 255) / 256, blk256, 0, stream>>>(h, out + (size_t)B_ * V_);
}
